// CrystalGraphE3ConvNet_15135464751901
// MI455X (gfx1250) — matrix operand order repaired and checked by reading, not compiled, not run on hardware
//
#include <hip/hip_runtime.h>

typedef __attribute__((ext_vector_type(16))) _Float16 v16h;
typedef __attribute__((ext_vector_type(8)))  float    v8f;

#define NATOMS 200000
#define MNBR   12
#define FDIM   64
#define NBRF   41
#define ORIGF  92
#define NCRYS  2000
#define APC    100
#define HF     128
#define NEDGE  (NATOMS * MNBR)

__device__ __forceinline__ float softplusf(float v) {
    return v > 20.0f ? v : log1pf(__expf(v));
}

__device__ __forceinline__ v8f wmma_f16(v16h a, v16h b, v8f c) {
    // (neg_a, A, neg_b, B, c_mod, C, reuse_a, reuse_b)
    return __builtin_amdgcn_wmma_f32_16x16x32_f16(false, a, false, b, (short)0, c,
                                                  false, false);
}

// A-frag K index: lanes0-15 hold K {0..7,16..23}, lanes16-31 hold {8..15,24..31}
__device__ __forceinline__ int a_k(int lane, int j) {
    return ((lane >> 4) << 3) + j + ((j >> 3) << 3);
}
// B-frag K index: lanes0-15 hold K {0..7,16..23}, lanes16-31 hold {8..15,24..31}
__device__ __forceinline__ int b_k(int lane, int j) {
    return ((lane >> 4) << 3) + j + ((j >> 3) << 3);
}

// ---------------------------------------------------------------------------
// Kernel 1: x = atom_fea(N,92) @ W_emb(92,64) + b_emb     [WMMA f16->f32]
// block = 256 thr = 8 waves; tile = 32 atoms x 64 cols; wave = 16x16 tile.
// atom_fea streamed with b128 loads; W_emb pre-marshaled into B-frag layout.
// ---------------------------------------------------------------------------
__global__ __launch_bounds__(256)
void k_embed(const float* __restrict__ atom_fea, const float* __restrict__ W_emb,
             const float* __restrict__ b_emb, float* __restrict__ x)
{
    __shared__ _Float16 ldsA[32 * 96];           // 6 KB
    __shared__ _Float16 ldsB[3 * 4 * 32 * 16];   // 12 KB, frag layout
    const int tid   = threadIdx.x;
    const int atom0 = blockIdx.x * 32;

    // A tile: 32*92 = 2944 floats = 736 float4 (base 32*92*4 B-aligned to 16)
    {
        const float4* src = (const float4*)(atom_fea + (size_t)atom0 * ORIGF);
        for (int i = tid; i < 736; i += 256) {
            float4 v = src[i];
            int fi = i * 4;
            int r0 = fi / ORIGF,       k0 = fi - r0 * ORIGF;
            int r1 = (fi + 1) / ORIGF, k1 = (fi + 1) - r1 * ORIGF;
            int r2 = (fi + 2) / ORIGF, k2 = (fi + 2) - r2 * ORIGF;
            int r3 = (fi + 3) / ORIGF, k3 = (fi + 3) - r3 * ORIGF;
            ldsA[r0 * 96 + k0] = (_Float16)v.x;
            ldsA[r1 * 96 + k1] = (_Float16)v.y;
            ldsA[r2 * 96 + k2] = (_Float16)v.z;
            ldsA[r3 * 96 + k3] = (_Float16)v.w;
        }
        for (int i = tid; i < 32 * 4; i += 256)          // zero pad k=92..95
            ldsA[(i >> 2) * 96 + ORIGF + (i & 3)] = (_Float16)0.0f;
    }
    for (int i = tid; i < 3 * 4 * 32 * 16; i += 256) {
        int j  = i & 15;
        int ln = (i >> 4) & 31;
        int nt = (i >> 9) & 3;
        int ks = i >> 11;
        int c  = nt * 16 + (ln & 15);
        int kb = ks * 32 + b_k(ln, j);
        ldsB[i] = (kb < ORIGF) ? (_Float16)W_emb[kb * FDIM + c] : (_Float16)0.0f;
    }
    __syncthreads();

    const int lane = tid & 31;
    const int wave = tid >> 5;
    const int mt   = wave >> 2;                    // 0..1
    const int nt   = wave & 3;                     // 0..3
    const int col  = nt * 16 + (lane & 15);
    const _Float16* aRow = &ldsA[(mt * 16 + (lane & 15)) * 96];
    v8f acc = {};
    #pragma unroll
    for (int ks = 0; ks < 3; ++ks) {
        v16h a, b;
        const _Float16* ap = aRow + ks * 32;
        const _Float16* bp = &ldsB[((ks * 4 + nt) * 32 + lane) * 16];
        #pragma unroll
        for (int j = 0; j < 16; ++j) { a[j] = ap[a_k(lane, j)]; b[j] = bp[j]; }
        acc = wmma_f16(a, b, acc);
    }
    const float bias = b_emb[col];
    #pragma unroll
    for (int v = 0; v < 8; ++v) {
        int row = v + ((lane >> 4) << 3);
        x[(atom0 + mt * 16 + row) * FDIM + col] = acc[v] + bias;
    }
}

// ---------------------------------------------------------------------------
// Kernel 2: edge scalars for ALL 3 layers in one pass over nbr_fea (394 MB).
//   coef[l][edge] = (softplus(e@Wr1[l]+br1[l]) . Wr2[l][:,0] + br2[l,0])
//                   * c0 * alpha / M
// block = 256 thr = 8 waves; 128 edges/block, 16 edges/wave.
// nbr_fea streamed with b128 loads into f16 LDS (K padded 41->64);
// Wr1 in LDS in B-frag layout; br1 / Wr2[:,0] zero-padded in LDS.
// ---------------------------------------------------------------------------
__global__ __launch_bounds__(256)
void k_edge_coef(const float* __restrict__ nbr_fea,
                 const float* __restrict__ Wr1, const float* __restrict__ br1,
                 const float* __restrict__ Wr2, const float* __restrict__ br2,
                 float* __restrict__ coef)
{
    __shared__ _Float16 ldsE[128 * 64];            // 16 KB
    __shared__ _Float16 ldsB[3 * 3 * 2 * 32 * 16]; // 18 KB, frag layout
    __shared__ float    ldsBias[3 * 48];
    __shared__ float    ldsW2[3 * 48];
    const int tid = threadIdx.x;
    const long long edgeBase = (long long)blockIdx.x * 128;

    // E tile: 128*41 = 5248 floats = 1312 float4 (base 128*41*4 B, 16-aligned)
    {
        const float4* src = (const float4*)(nbr_fea + edgeBase * NBRF);
        for (int i = tid; i < 1312; i += 256) {
            float4 v = src[i];
            int fi = i * 4;
            int r0 = fi / NBRF,       k0 = fi - r0 * NBRF;
            int r1 = (fi + 1) / NBRF, k1 = (fi + 1) - r1 * NBRF;
            int r2 = (fi + 2) / NBRF, k2 = (fi + 2) - r2 * NBRF;
            int r3 = (fi + 3) / NBRF, k3 = (fi + 3) - r3 * NBRF;
            ldsE[r0 * 64 + k0] = (_Float16)v.x;
            ldsE[r1 * 64 + k1] = (_Float16)v.y;
            ldsE[r2 * 64 + k2] = (_Float16)v.z;
            ldsE[r3 * 64 + k3] = (_Float16)v.w;
        }
        for (int i = tid; i < 128 * 23; i += 256) {      // zero pad k=41..63
            int r = i / 23, k = NBRF + (i - r * 23);
            ldsE[r * 64 + k] = (_Float16)0.0f;
        }
    }
    for (int i = tid; i < 3 * 3 * 2 * 32 * 16; i += 256) {
        int idx = i;
        int j = idx & 15;      idx >>= 4;
        int ln = idx & 31;     idx >>= 5;
        int ks = idx & 1;      idx >>= 1;
        int nt = idx % 3;
        int l  = idx / 3;
        int c  = nt * 16 + (ln & 15);
        int kb = ks * 32 + b_k(ln, j);
        ldsB[i] = (c < NBRF && kb < NBRF)
                    ? (_Float16)Wr1[l * NBRF * NBRF + kb * NBRF + c]
                    : (_Float16)0.0f;
    }
    for (int i = tid; i < 3 * 48; i += 256) {
        int l = i / 48, c = i % 48;
        ldsBias[i] = (c < NBRF) ? br1[l * NBRF + c] : 0.0f;
        ldsW2[i]   = (c < NBRF) ? Wr2[(l * NBRF + c) * 9] : 0.0f; // column 0
    }
    __syncthreads();

    const int lane = tid & 31;
    const int wave = tid >> 5;                 // 16 edges each
    const _Float16* eT = &ldsE[wave * 16 * 64];
    v16h a0, a1;
    #pragma unroll
    for (int j = 0; j < 16; ++j) {
        int ka = a_k(lane, j);
        a0[j] = eT[(lane & 15) * 64 + ka];
        a1[j] = eT[(lane & 15) * 64 + 32 + ka];
    }

    const float SCALE = 0.28209479177387814f * 0.125f / 12.0f; // c0*alpha/M
    #pragma unroll
    for (int l = 0; l < 3; ++l) {
        float sums[8];
        #pragma unroll
        for (int v = 0; v < 8; ++v) sums[v] = 0.0f;

        #pragma unroll
        for (int nt = 0; nt < 3; ++nt) {
            const int c = nt * 16 + (lane & 15);
            v16h b0, b1;
            const _Float16* bp0 = &ldsB[(((l * 3 + nt) * 2 + 0) * 32 + lane) * 16];
            const _Float16* bp1 = &ldsB[(((l * 3 + nt) * 2 + 1) * 32 + lane) * 16];
            #pragma unroll
            for (int j = 0; j < 16; ++j) { b0[j] = bp0[j]; b1[j] = bp1[j]; }
            v8f acc = {};
            acc = wmma_f16(a0, b0, acc);
            acc = wmma_f16(a1, b1, acc);
            const float bias = ldsBias[l * 48 + c];
            const float w2   = ldsW2[l * 48 + c];   // 0 for padded columns
            #pragma unroll
            for (int v = 0; v < 8; ++v)
                sums[v] += softplusf(acc[v] + bias) * w2;
        }
        // Reduce over columns: C layout => sum across the 16-lane half.
        float val = 0.0f;
        #pragma unroll
        for (int v = 0; v < 8; ++v) {
            float t = sums[v];
            t += __shfl_xor(t, 1);
            t += __shfl_xor(t, 2);
            t += __shfl_xor(t, 4);
            t += __shfl_xor(t, 8);
            if ((lane & 7) == v) val = t;   // lanes0-15: row v; lanes16-31: row v+8
        }
        if ((lane & 15) < 8) {
            int row = (lane < 16) ? lane : lane - 8;
            coef[(long long)l * NEDGE + edgeBase + wave * 16 + row] =
                (val + br2[l * 9]) * SCALE;
        }
    }
}

// ---------------------------------------------------------------------------
// Kernel 3 (per layer): weighted neighbor gather + GEMM with Wtp[l].
//   agg[i] = sum_m coef[i,m] * x[src[i,m]]  (f32 gather, f16 stage in LDS)
//   xout   = agg(32x64) @ Wtp(64x64)        [WMMA, Wtp staged in frag layout]
// block = 256 thr = 8 waves; 32 atoms/block; wave = one 16x16 output tile.
// nbr_idx / coef read as int4/float4 (48 B/atom, 16-aligned).
// ---------------------------------------------------------------------------
__global__ __launch_bounds__(256)
void k_conv(const float* __restrict__ xin, const int* __restrict__ nbr_idx,
            const float* __restrict__ coef_l, const float* __restrict__ Wtp_l,
            float* __restrict__ xout)
{
    __shared__ _Float16 ldsA[32 * 64];           // 4 KB
    __shared__ _Float16 ldsB[2 * 4 * 32 * 16];   // 8 KB, frag layout
    const int tid   = threadIdx.x;
    const int atom0 = blockIdx.x * 32;
    const int f     = tid & 63;
    const int sub   = tid >> 6;         // 0..3
    for (int a = sub; a < 32; a += 4) {
        const long long eb = (long long)(atom0 + a) * MNBR;
        const int4*   ni = (const int4*)(nbr_idx + eb);
        const float4* nc = (const float4*)(coef_l + eb);
        float acc = 0.0f;
        #pragma unroll
        for (int q = 0; q < 3; ++q) {
            int4   s4 = ni[q];
            float4 c4 = nc[q];
            acc += c4.x * xin[(long long)s4.x * FDIM + f];
            acc += c4.y * xin[(long long)s4.y * FDIM + f];
            acc += c4.z * xin[(long long)s4.z * FDIM + f];
            acc += c4.w * xin[(long long)s4.w * FDIM + f];
        }
        ldsA[a * 64 + f] = (_Float16)acc;
    }
    for (int i = tid; i < 2 * 4 * 32 * 16; i += 256) {
        int j  = i & 15;
        int ln = (i >> 4) & 31;
        int nt = (i >> 9) & 3;
        int ks = i >> 11;
        ldsB[i] = (_Float16)Wtp_l[(ks * 32 + b_k(ln, j)) * FDIM + nt * 16 + (ln & 15)];
    }
    __syncthreads();

    const int lane = tid & 31;
    const int wave = tid >> 5;
    const int mt   = wave >> 2;                    // 0..1 row tile
    const int nt   = wave & 3;
    const int col  = nt * 16 + (lane & 15);
    const _Float16* aT = &ldsA[(mt * 16 + (lane & 15)) * 64];
    v8f acc = {};
    #pragma unroll
    for (int ks = 0; ks < 2; ++ks) {
        v16h a, b;
        const _Float16* ap = aT + ks * 32;
        const _Float16* bp = &ldsB[((ks * 4 + nt) * 32 + lane) * 16];
        #pragma unroll
        for (int j = 0; j < 16; ++j) { a[j] = ap[a_k(lane, j)]; b[j] = bp[j]; }
        acc = wmma_f16(a, b, acc);
    }
    #pragma unroll
    for (int v = 0; v < 8; ++v) {
        int row = v + ((lane >> 4) << 3);
        xout[(long long)(atom0 + mt * 16 + row) * FDIM + col] = acc[v];
    }
}

// ---------------------------------------------------------------------------
// Kernel 4: readout. crys = mean over 100 atoms; h = softplus(crys@W_fc+b_fc);
// out = h@W_out + b_out. d_out layout: out[2000] then h[2000*128].
// one block (128 thr) per crystal.
// ---------------------------------------------------------------------------
__global__ __launch_bounds__(128)
void k_readout(const float* __restrict__ x, const int* __restrict__ cidx,
               const float* __restrict__ W_fc, const float* __restrict__ b_fc,
               const float* __restrict__ W_out, const float* __restrict__ b_out,
               float* __restrict__ out)
{
    __shared__ float cf[64];
    __shared__ float red[128];
    const int b    = blockIdx.x;
    const int tid  = threadIdx.x;
    const int f    = tid & 63;
    const int half = tid >> 6;
    float s = 0.0f;
    for (int a = half; a < APC; a += 2) {
        int atom = cidx[b * APC + a];
        s += x[(long long)atom * FDIM + f];
    }
    red[tid] = s;
    __syncthreads();
    if (tid < 64) cf[tid] = (red[tid] + red[tid + 64]) * (1.0f / APC);
    __syncthreads();

    float hv = b_fc[tid];
    #pragma unroll
    for (int k = 0; k < 64; ++k) hv += cf[k] * W_fc[k * HF + tid];
    hv = softplusf(hv);
    out[NCRYS + (long long)b * HF + tid] = hv;
    red[tid] = hv * W_out[tid];
    __syncthreads();
    #pragma unroll
    for (int st = 64; st > 0; st >>= 1) {
        if (tid < st) red[tid] += red[tid + st];
        __syncthreads();
    }
    if (tid == 0) out[b] = red[0] + b_out[0];
}

// ---------------------------------------------------------------------------
extern "C" void kernel_launch(void* const* d_in, const int* in_sizes, int n_in,
                              void* d_out, int out_size, void* d_ws, size_t ws_size,
                              hipStream_t stream)
{
    (void)in_sizes; (void)n_in; (void)out_size; (void)ws_size;
    const float* atom_fea = (const float*)d_in[0];
    const float* nbr_fea  = (const float*)d_in[1];
    const int*   nbr_idx  = (const int*)d_in[2];
    const int*   cidx     = (const int*)d_in[3];
    // d_in[4] (pos) unused: only the constant l=0 harmonic Y0 survives.
    const float* W_emb = (const float*)d_in[5];
    const float* b_emb = (const float*)d_in[6];
    const float* Wr1   = (const float*)d_in[7];
    const float* br1   = (const float*)d_in[8];
    const float* Wr2   = (const float*)d_in[9];
    const float* br2   = (const float*)d_in[10];
    const float* Wtp   = (const float*)d_in[11];
    const float* W_fc  = (const float*)d_in[12];
    const float* b_fc  = (const float*)d_in[13];
    const float* W_out = (const float*)d_in[14];
    const float* b_out = (const float*)d_in[15];
    float* out = (float*)d_out;

    char*  ws   = (char*)d_ws;
    float* xA   = (float*)ws;                                    // N*64 f32
    float* xB   = (float*)(ws + (size_t)NATOMS * FDIM * 4);      // N*64 f32
    float* coef = (float*)(ws + (size_t)NATOMS * FDIM * 4 * 2);  // 3*NEDGE f32

    k_embed<<<NATOMS / 32, 256, 0, stream>>>(atom_fea, W_emb, b_emb, xA);
    k_edge_coef<<<NEDGE / 128, 256, 0, stream>>>(nbr_fea, Wr1, br1, Wr2, br2, coef);

    float* cur = xA;
    float* nxt = xB;
    for (int l = 0; l < 3; ++l) {
        k_conv<<<NATOMS / 32, 256, 0, stream>>>(cur, nbr_idx,
                                                coef + (size_t)l * NEDGE,
                                                Wtp + (size_t)l * FDIM * FDIM, nxt);
        float* t = nxt; nxt = cur; cur = t;
    }
    k_readout<<<NCRYS, 128, 0, stream>>>(cur, cidx, W_fc, b_fc, W_out, b_out, out);
}
